// Multi_Head_Latent_Attention_3951369912885
// MI455X (gfx1250) — compile-verified, matrix operand order checked by reading, not run on hardware
//
#include <hip/hip_runtime.h>

typedef unsigned short u16t;
typedef __attribute__((ext_vector_type(16))) __bf16 v16bf;
typedef __attribute__((ext_vector_type(8)))  float  v8f;
typedef __attribute__((ext_vector_type(4)))  unsigned tdm4u;
typedef __attribute__((ext_vector_type(8)))  int      tdm8i;
typedef __attribute__((ext_vector_type(4)))  int      tdm4i;

#define HIDDEN_ 2048
#define HEADS_  16
#define KV_LORA_ 512
#define ROPE_D_ 64
#define QHD_    128   // q head dim (rope+nope)
#define VD_     128
#define BB_     2
#define SS_     2048
#define SCALE_  0.125f  // 64^-0.5

#if __has_builtin(__builtin_amdgcn_tensor_load_to_lds) && __has_builtin(__builtin_amdgcn_s_wait_tensorcnt)
#define USE_TDM 1
#endif

__device__ __forceinline__ u16t f2bf(float f) {
  unsigned u = __float_as_uint(f);
  u += 0x7FFFu + ((u >> 16) & 1u);   // round-to-nearest-even
  return (u16t)(u >> 16);
}

struct Frag32B { uint4 lo, hi; };

__device__ __forceinline__ v16bf ld_frag(const u16t* plo, const u16t* phi) {
  Frag32B t;
  t.lo = *(const uint4*)plo;   // fragment elems 0..7  (K = kh*8 .. kh*8+7)
  t.hi = *(const uint4*)phi;   // fragment elems 8..15 (K = 16+kh*8 ..)
  return __builtin_bit_cast(v16bf, t);
}

#ifdef USE_TDM
// TDM 2D tile load: rows x rowlen (bf16 elems), global row stride = stride elems,
// into contiguous LDS at lds_off. Descriptor per CDNA5 ISA.
// 6-arg builtin form (clang-23 / therock-10.0): (g0 v4u, g1 v8i, v4i, v4i, v8i, cpol)
__device__ __forceinline__ void tdm_load_2d(const u16t* gsrc, unsigned lds_off,
                                            unsigned rows, unsigned rowlen,
                                            unsigned stride) {
  unsigned long long ga = (unsigned long long)(uintptr_t)gsrc;
  // group0: count=1 [1:0], lds_addr [63:32], global_addr [120:64], type=2 [127:126]
  tdm4u g0 = { 1u, lds_off, (unsigned)ga,
               (unsigned)((ga >> 32) & 0x1FFFFFFu) | 0x80000000u };
  unsigned td1 = 1u << 20;   // tensor_dim1: large (tiles always in-bounds)
  // group1: data_size=1(2B) [17:16]; tensor_dim0 [79:48]; tensor_dim1 [111:80];
  // tile_dim0 [127:112]; tile_dim1 [143:128]; tensor_dim0_stride [207:160]
  tdm8i g1 = { (int)(1u << 16),
               (int)((stride & 0xffffu) << 16),
               (int)(((stride >> 16) & 0xffffu) | ((td1 & 0xffffu) << 16)),
               (int)(((td1 >> 16) & 0xffffu) | (rowlen << 16)),
               (int)(rows & 0xffffu),
               (int)stride,
               0, 0 };
  tdm4i gz4 = { 0, 0, 0, 0 };
  tdm8i gz8 = { 0, 0, 0, 0, 0, 0, 0, 0 };
  __builtin_amdgcn_tensor_load_to_lds(g0, g1, gz4, gz4, gz8, 0);
}
#endif

__global__ void __launch_bounds__(256)
f32_to_bf16(const float* __restrict__ in, u16t* __restrict__ out, int n) {
  int i = blockIdx.x * 256 + threadIdx.x;
  if (i < n) out[i] = f2bf(in[i]);
}

// vt[((b*H+h)*VD + d)*S + key] = kvup[b, key, h*192 + 64 + d]
__global__ void __launch_bounds__(256)
transpose_v(const u16t* __restrict__ kvup, u16t* __restrict__ vt, int total) {
  int i = blockIdx.x * 256 + threadIdx.x;
  if (i >= total) return;
  int key = i & (SS_ - 1);
  int d   = (i >> 11) & (VD_ - 1);
  int hh  = (i >> 18) & (HEADS_ - 1);
  int bb  = i >> 22;
  vt[i] = kvup[((size_t)bb * SS_ + key) * (HEADS_ * 192) + hh * 192 + 64 + d];
}

// C[M,N] = A[M,K] * B[K,N]; bf16 in, f32 accumulate, bf16 or f32 out.
// 256 threads = 8 waves; block tile 128(M) x 64(N); wave = 32x32 (4 accumulators).
template<bool OUT_BF16>
__global__ void __launch_bounds__(256)
gemm_bf16(const u16t* __restrict__ A, const u16t* __restrict__ Bm,
          void* __restrict__ C, int M, int N, int K,
          int lda, int ldb, int ldc) {
  __shared__ alignas(16) u16t As[128 * 32];   // [m][k]   8KB
  __shared__ alignas(16) u16t Bst[64 * 32];   // [n][k]   4KB (transposed)
  const int tid  = threadIdx.x;
  const int wave = tid >> 5, lane = tid & 31;
  const int wm = wave & 3, wn = wave >> 2;
  const int nl = lane & 15, kh = lane >> 4;
  const int m0 = blockIdx.y * 128, n0 = blockIdx.x * 64;
  v8f acc00 = {}, acc01 = {}, acc10 = {}, acc11 = {};
  for (int k0 = 0; k0 < K; k0 += 32) {
    // ---- stage A tile (128x32) ----
#ifdef USE_TDM
    if (wave == 0) {
      tdm_load_2d(&A[(size_t)m0 * lda + k0], (unsigned)(uintptr_t)As,
                  128u, 32u, (unsigned)lda);
      __builtin_amdgcn_s_wait_tensorcnt(0);
    }
#else
#pragma unroll
    for (int i = 0; i < 16; ++i) {
      int idx = tid + i * 256;
      int r = idx >> 5, c = idx & 31;
      As[idx] = A[(size_t)(m0 + r) * lda + (k0 + c)];
    }
#endif
    // ---- stage B tile (32k x 64n), transposed into LDS ----
#pragma unroll
    for (int i = 0; i < 8; ++i) {
      int idx = tid + i * 256;
      int r = idx >> 6, c = idx & 63;
      Bst[c * 32 + r] = Bm[(size_t)(k0 + r) * ldb + (n0 + c)];
    }
    if (k0 + 32 < K) {  // prefetch next B panel row for this thread's first chunk
      __builtin_prefetch(&Bm[(size_t)(k0 + 32 + (tid >> 6)) * ldb + n0 + (tid & 63)], 0, 1);
    }
    __syncthreads();
    const u16t* pa0 = &As[(wm * 32 + nl) * 32 + kh * 8];
    const u16t* pa1 = pa0 + 16 * 32;
    v16bf a0 = ld_frag(pa0, pa0 + 16);
    v16bf a1 = ld_frag(pa1, pa1 + 16);
    const u16t* pb0 = &Bst[(wn * 32 + nl) * 32 + kh * 8];
    const u16t* pb1 = pb0 + 16 * 32;
    v16bf b0 = ld_frag(pb0, pb0 + 16);
    v16bf b1 = ld_frag(pb1, pb1 + 16);
    acc00 = __builtin_amdgcn_wmma_f32_16x16x32_bf16(false, a0, false, b0, (short)0, acc00, false, false);
    acc01 = __builtin_amdgcn_wmma_f32_16x16x32_bf16(false, a0, false, b1, (short)0, acc01, false, false);
    acc10 = __builtin_amdgcn_wmma_f32_16x16x32_bf16(false, a1, false, b0, (short)0, acc10, false, false);
    acc11 = __builtin_amdgcn_wmma_f32_16x16x32_bf16(false, a1, false, b1, (short)0, acc11, false, false);
    __syncthreads();
  }
#pragma unroll
  for (int v = 0; v < 8; ++v) {
    int mmA = m0 + wm * 32 + kh * 8 + v;        // rows of a0-tiles
    int mmB = mmA + 16;                          // rows of a1-tiles
    int nn0 = n0 + wn * 32 + nl;
    int nn1 = nn0 + 16;
    if (OUT_BF16) {
      u16t* Cb = (u16t*)C;
      Cb[(size_t)mmA * ldc + nn0] = f2bf(acc00[v]);
      Cb[(size_t)mmA * ldc + nn1] = f2bf(acc01[v]);
      Cb[(size_t)mmB * ldc + nn0] = f2bf(acc10[v]);
      Cb[(size_t)mmB * ldc + nn1] = f2bf(acc11[v]);
    } else {
      float* Cf = (float*)C;
      Cf[(size_t)mmA * ldc + nn0] = acc00[v];
      Cf[(size_t)mmA * ldc + nn1] = acc01[v];
      Cf[(size_t)mmB * ldc + nn0] = acc10[v];
      Cf[(size_t)mmB * ldc + nn1] = acc11[v];
    }
  }
}

// Flash-attention: one wave per (b, h, 16-row q tile). 32 keys per iteration.
__global__ void __launch_bounds__(256)
mla_attn(const u16t* __restrict__ qb, const u16t* __restrict__ kvdown,
         const u16t* __restrict__ kvup, const u16t* __restrict__ vt,
         const float* __restrict__ mask, u16t* __restrict__ ctx) {
  __shared__ alignas(16) u16t Ps_all[8 * 16 * 32];   // per-wave P scratch (16 rows x 32 keys)
  const int wave = threadIdx.x >> 5;
  const int lane = threadIdx.x & 31;
  u16t* Ps = &Ps_all[wave * 16 * 32];
  const int gw = blockIdx.x * 8 + wave;
  const int qt = gw & 127;            // S/16 = 128 q tiles
  const int h  = (gw >> 7) & 15;
  const int b  = gw >> 11;
  const int q0 = qt * 16;
  const int nl = lane & 15, kh = lane >> 4;

  // Q fragments: 4 d-blocks of 32, loaded once (q_full == q)
  v16bf qa[4];
#pragma unroll
  for (int d = 0; d < 4; ++d) {
    const u16t* p = &qb[((size_t)b * SS_ + q0 + nl) * (HEADS_ * QHD_) + h * QHD_ + d * 32 + kh * 8];
    qa[d] = ld_frag(p, p + 16);
  }

  float rmax[8], rsum[8];
  v8f acc[8];
  v8f zero = {};
#pragma unroll
  for (int v = 0; v < 8; ++v) { rmax[v] = -1e30f; rsum[v] = 0.f; }
#pragma unroll
  for (int dt = 0; dt < 8; ++dt) acc[dt] = zero;

  for (int kb = 0; kb < SS_; kb += 32) {
    v8f s0 = {}, s1 = {};
#pragma unroll
    for (int d = 0; d < 4; ++d) {     // d<2: rope dims (kv_down cols 512..), d>=2: nope (kv_up)
      int dd = d * 32 + kh * 8;
      const u16t *pk0, *pk1;
      if (d < 2) {
        pk0 = &kvdown[((size_t)b * SS_ + kb + nl) * 576 + 512 + dd];
        pk1 = &kvdown[((size_t)b * SS_ + kb + 16 + nl) * 576 + 512 + dd];
      } else {
        pk0 = &kvup[((size_t)b * SS_ + kb + nl) * (HEADS_ * 192) + h * 192 + (dd - 64)];
        pk1 = &kvup[((size_t)b * SS_ + kb + 16 + nl) * (HEADS_ * 192) + h * 192 + (dd - 64)];
      }
      v16bf k0f = ld_frag(pk0, pk0 + 16);
      v16bf k1f = ld_frag(pk1, pk1 + 16);
      s0 = __builtin_amdgcn_wmma_f32_16x16x32_bf16(false, qa[d], false, k0f, (short)0, s0, false, false);
      s1 = __builtin_amdgcn_wmma_f32_16x16x32_bf16(false, qa[d], false, k1f, (short)0, s1, false, false);
    }
    // Online softmax. D-layout: lane holds key n = nl (s0) / nl+16 (s1); row = v + 8*kh.
#pragma unroll
    for (int v = 0; v < 8; ++v) {
      int qrow = q0 + v + 8 * kh;
      float x0 = s0[v] * SCALE_ + mask[(size_t)qrow * SS_ + kb + nl];
      float x1 = s1[v] * SCALE_ + mask[(size_t)qrow * SS_ + kb + 16 + nl];
      float mx = fmaxf(x0, x1);
#pragma unroll
      for (int off = 1; off < 16; off <<= 1)
        mx = fmaxf(mx, __shfl_xor(mx, off, 32));
      float nm = fmaxf(rmax[v], mx);
      float sc = __expf(rmax[v] - nm);
      float p0 = __expf(x0 - nm);
      float p1 = __expf(x1 - nm);
      float ps = p0 + p1;
#pragma unroll
      for (int off = 1; off < 16; off <<= 1)
        ps += __shfl_xor(ps, off, 32);
      rsum[v] = rsum[v] * sc + ps;
      rmax[v] = nm;
#pragma unroll
      for (int dt = 0; dt < 8; ++dt) acc[dt][v] = acc[dt][v] * sc;
      Ps[(v + 8 * kh) * 32 + nl]      = f2bf(p0);
      Ps[(v + 8 * kh) * 32 + 16 + nl] = f2bf(p1);
    }
    __syncthreads();
    // P (16x32) as A fragment; V^T rows give contiguous-key B fragments.
    const u16t* pp = &Ps[nl * 32 + kh * 8];
    v16bf pfrag = ld_frag(pp, pp + 16);
#pragma unroll
    for (int dt = 0; dt < 8; ++dt) {
      const u16t* pv = &vt[(((size_t)b * HEADS_ + h) * VD_ + dt * 16 + nl) * SS_ + kb + kh * 8];
      v16bf vfrag = ld_frag(pv, pv + 16);
      acc[dt] = __builtin_amdgcn_wmma_f32_16x16x32_bf16(false, pfrag, false, vfrag, (short)0, acc[dt], false, false);
    }
    __syncthreads();
  }
#pragma unroll
  for (int dt = 0; dt < 8; ++dt) {
#pragma unroll
    for (int v = 0; v < 8; ++v) {
      int qrow = q0 + v + 8 * kh;
      ctx[((size_t)b * SS_ + qrow) * (HEADS_ * VD_) + h * VD_ + dt * 16 + nl] =
          f2bf(acc[dt][v] / rsum[v]);
    }
  }
}

extern "C" void kernel_launch(void* const* d_in, const int* in_sizes, int n_in,
                              void* d_out, int out_size, void* d_ws, size_t ws_size,
                              hipStream_t stream) {
  const float* hidden = (const float*)d_in[0];
  const float* mask   = (const float*)d_in[1];
  const float* Wdown  = (const float*)d_in[2];
  const float* Wup    = (const float*)d_in[3];
  const float* Wq     = (const float*)d_in[4];
  const float* Wo     = (const float*)d_in[5];
  float* out = (float*)d_out;

  const size_t n_hidden = (size_t)BB_ * SS_ * HIDDEN_;         // 8388608
  const size_t n_wdown  = (size_t)HIDDEN_ * 576;
  const size_t n_wup    = (size_t)KV_LORA_ * 3072;
  const size_t n_wq     = (size_t)HIDDEN_ * 2048;
  const size_t n_wo     = (size_t)2048 * HIDDEN_;
  const size_t n_kvdown = (size_t)BB_ * SS_ * 576;
  const size_t n_kvup   = (size_t)BB_ * SS_ * 3072;
  const size_t n_q      = (size_t)BB_ * SS_ * 2048;
  const size_t n_vt     = (size_t)BB_ * HEADS_ * VD_ * SS_;
  const size_t n_ctx    = (size_t)BB_ * SS_ * 2048;

  u16t* ws   = (u16t*)d_ws;
  u16t* h_bf  = ws;  ws += n_hidden;
  u16t* wd_bf = ws;  ws += n_wdown;
  u16t* wu_bf = ws;  ws += n_wup;
  u16t* wq_bf = ws;  ws += n_wq;
  u16t* wo_bf = ws;  ws += n_wo;
  u16t* kvd   = ws;  ws += n_kvdown;
  u16t* kvu   = ws;  ws += n_kvup;
  u16t* qb    = ws;  ws += n_q;
  u16t* vt    = ws;  ws += n_vt;
  u16t* ctx   = ws;  ws += n_ctx;

  auto cv = [&](const float* src, u16t* dst, size_t n) {
    f32_to_bf16<<<dim3((unsigned)((n + 255) / 256)), dim3(256), 0, stream>>>(src, dst, (int)n);
  };
  cv(hidden, h_bf, n_hidden);
  cv(Wdown, wd_bf, n_wdown);
  cv(Wup,   wu_bf, n_wup);
  cv(Wq,    wq_bf, n_wq);
  cv(Wo,    wo_bf, n_wo);

  // kv_down = hidden @ W_down : 4096x576, K=2048
  gemm_bf16<true><<<dim3(576 / 64, 4096 / 128), 256, 0, stream>>>(
      h_bf, wd_bf, kvd, 4096, 576, 2048, 2048, 576, 576);
  // kv_up = kv_compressed @ W_up : 4096x3072, K=512 (A row stride 576)
  gemm_bf16<true><<<dim3(3072 / 64, 4096 / 128), 256, 0, stream>>>(
      kvd, wu_bf, kvu, 4096, 3072, 512, 576, 3072, 3072);
  // q = hidden @ W_q : 4096x2048, K=2048
  gemm_bf16<true><<<dim3(2048 / 64, 4096 / 128), 256, 0, stream>>>(
      h_bf, wq_bf, qb, 4096, 2048, 2048, 2048, 2048, 2048);
  // V transpose to [b,h,d,key]
  transpose_v<<<dim3((unsigned)(n_vt / 256)), 256, 0, stream>>>(kvu, vt, (int)n_vt);
  // attention: 4096 waves / 8 per block
  mla_attn<<<dim3(512), 256, 0, stream>>>(qb, kvd, kvu, vt, mask, ctx);
  // out = ctx @ W_o : 4096x2048, K=2048, fp32 output
  gemm_bf16<false><<<dim3(2048 / 64, 4096 / 128), 256, 0, stream>>>(
      ctx, wo_bf, out, 4096, 2048, 2048, 2048, 2048, 2048);
}
